// GLCM_64510408786136
// MI455X (gfx1250) — 8-bit matrix operand order repaired and checked by reading, compiled, not run on hardware
//
#include <hip/hip_runtime.h>

typedef __attribute__((ext_vector_type(16))) int   v16i_t;
typedef __attribute__((ext_vector_type(8)))  float v8f_t;

#define B_  16
#define H_  384
#define W_  384
#define C_  3

__device__ __forceinline__ unsigned bin4(float x) {
    return (unsigned)(int)(x * 2.66666675f);
}

__device__ __forceinline__ unsigned oh4(unsigned v, unsigned sh) {
    return (v >> sh) & 0x01010101u;
}

__device__ __forceinline__ void build_frag128(const unsigned* __restrict__ lds32,
                                              unsigned wbase, int t, int c,
                                              bool hh, unsigned sh,
                                              v16i_t& Af, v16i_t& Bf) {
    const unsigned baseB = wbase + (unsigned)c * 128u + (unsigned)t * 32u;
    const unsigned baseS = wbase + 384u + (unsigned)c * 128u + (unsigned)t * 32u;

    const unsigned ha = hh ? 2u : 0u;
    #pragma unroll
    for (int half = 0; half < 2; ++half) {
        #pragma unroll
        for (int pp = 0; pp < 4; ++pp) {
            uint2 u = *(const uint2*)&lds32[baseB + (unsigned)(half * 16 + (pp >> 1) * 8 + (pp & 1) * 2) + 2u * ha];
            Af[half * 8 + 2 * pp]     = (int)oh4(u.x, sh);
            Af[half * 8 + 2 * pp + 1] = (int)oh4(u.y, sh);
        }
    }
    const unsigned hb = hh ? 4u : 0u;
    #pragma unroll
    for (int q = 0; q < 4; ++q) {
        uint4 u = *(const uint4*)&lds32[baseS + (unsigned)(q * 8) + hb];
        Bf[4 * q + 0] = (int)oh4(u.x, sh);
        Bf[4 * q + 1] = (int)oh4(u.y, sh);
        Bf[4 * q + 2] = (int)oh4(u.z, sh);
        Bf[4 * q + 3] = (int)oh4(u.w, sh);
    }
}

__global__ __launch_bounds__(256) void glcm_zero(unsigned* __restrict__ out) {
    out[blockIdx.x * 256 + threadIdx.x] = 0u;
}

__global__ __launch_bounds__(256) void glcm_main(const float* __restrict__ in,
                                                 unsigned* __restrict__ out) {
    __shared__ unsigned lds32[8 * 768];

    const int  tid  = threadIdx.x;
    const int  lane = tid & 31;
    const int  wv   = tid >> 5;
    const int  b    = blockIdx.x >> 3;
    const int  oct  = blockIdx.x & 7;
    const unsigned wbase = (unsigned)wv * 768u;
    const bool hh   = lane >= 16;
    const unsigned sh = (unsigned)(lane & 15);

    if (lane < 3) lds32[wbase + (unsigned)lane * 128u + 96u] = 0u;

    v8f_t acc0 = {}, acc1 = {}, acc2 = {};

    for (int r = 0; r < 6; ++r) {
        const int h = oct * 48 + wv * 6 + r;
        const float* rowp = in + (size_t)(b * H_ + h) * (W_ * C_);

        __syncthreads();

        #pragma unroll
        for (int p = 0; p < 3; ++p) {
            const float* fp = rowp + p * 384 + lane * 12;
            float4 f0 = *(const float4*)(fp);
            float4 f1 = *(const float4*)(fp + 4);
            float4 f2 = *(const float4*)(fp + 8);
            unsigned d0 = (1u << bin4(f0.x)) | ((1u << bin4(f0.w)) << 8) |
                          ((1u << bin4(f1.z)) << 16) | ((1u << bin4(f2.y)) << 24);
            unsigned d1 = (1u << bin4(f0.y)) | ((1u << bin4(f1.x)) << 8) |
                          ((1u << bin4(f1.w)) << 16) | ((1u << bin4(f2.z)) << 24);
            unsigned d2 = (1u << bin4(f0.z)) | ((1u << bin4(f1.y)) << 8) |
                          ((1u << bin4(f2.x)) << 16) | ((1u << bin4(f2.w)) << 24);
            lds32[wbase + 0u * 128u + (unsigned)(p * 32 + lane)] = d0;
            lds32[wbase + 1u * 128u + (unsigned)(p * 32 + lane)] = d1;
            lds32[wbase + 2u * 128u + (unsigned)(p * 32 + lane)] = d2;
        }
        __syncthreads();

        #pragma unroll
        for (int c = 0; c < 3; ++c) {
            #pragma unroll
            for (int s = 0; s < 3; ++s) {
                const unsigned d  = (unsigned)(s * 32 + lane);
                unsigned lo = lds32[wbase + (unsigned)c * 128u + d];
                unsigned hi = lds32[wbase + (unsigned)c * 128u + d + 1u];
                lds32[wbase + 384u + (unsigned)c * 128u + d] = (lo >> 8) | (hi << 24);
            }
        }
        __syncthreads();

        #pragma unroll
        for (int t = 0; t < 3; ++t) {
            #pragma unroll
            for (int c = 0; c < 3; ++c) {
                v16i_t Af, Bf;
                build_frag128(lds32, wbase, t, c, hh, sh, Af, Bf);
                if (c == 0)
                    acc0 = __builtin_amdgcn_wmma_f32_16x16x128_fp8_fp8(Af, Bf, (short)0, acc0, false, false);
                else if (c == 1)
                    acc1 = __builtin_amdgcn_wmma_f32_16x16x128_fp8_fp8(Af, Bf, (short)0, acc1, false, false);
                else
                    acc2 = __builtin_amdgcn_wmma_f32_16x16x128_fp8_fp8(Af, Bf, (short)0, acc2, false, false);
            }
        }
    }

    __syncthreads();
    {
        const int col = lane & 15;
        const int rb  = hh ? 8 : 0;
        #pragma unroll
        for (int rr = 0; rr < 8; ++rr) {
            lds32[wbase + 0u * 256u + (unsigned)((rb + rr) * 16 + col)] = (unsigned)(acc0[rr] * 262144.0f);
            lds32[wbase + 1u * 256u + (unsigned)((rb + rr) * 16 + col)] = (unsigned)(acc1[rr] * 262144.0f);
            lds32[wbase + 2u * 256u + (unsigned)((rb + rr) * 16 + col)] = (unsigned)(acc2[rr] * 262144.0f);
        }
    }
    __syncthreads();

    #pragma unroll
    for (int c = 0; c < 3; ++c) {
        #pragma unroll
        for (int u = 0; u < 2; ++u) {
            const int idx = u * 32 + lane;
            const int i = idx >> 3, j = idx & 7;
            unsigned mij = lds32[wbase + (unsigned)c * 256u + (unsigned)(i * 16 + j)];
            unsigned mji = lds32[wbase + (unsigned)c * 256u + (unsigned)(j * 16 + i)];
            atomicAdd(&out[b * 192 + c * 64 + idx], mij + mji);
        }
    }
}

__global__ __launch_bounds__(256) void glcm_final(unsigned* __restrict__ io) {
    const int k = blockIdx.x * 256 + threadIdx.x;
    const float inv = 1.0f / 14118912.0f;
    float v = (float)io[k] * inv;
    ((float*)io)[k] = v;
}

extern "C" void kernel_launch(void* const* d_in, const int* in_sizes, int n_in,
                              void* d_out, int out_size, void* d_ws, size_t ws_size,
                              hipStream_t stream) {
    (void)in_sizes; (void)n_in; (void)d_ws; (void)ws_size; (void)out_size;
    const float* in = (const float*)d_in[0];
    unsigned* outw = (unsigned*)d_out;
    glcm_zero <<<12, 256, 0, stream>>>(outw);
    glcm_main <<<128, 256, 0, stream>>>(in, outw);
    glcm_final<<<12, 256, 0, stream>>>(outw);
}
